// Trend_26998164422875
// MI455X (gfx1250) — matrix operand order repaired and checked by reading, not compiled, not run on hardware
//
#include <hip/hip_runtime.h>
#include <hip/hip_bf16.h>
#include <math.h>

// ---------------------------------------------------------------------------
// Trend attention:  LN -> proj (WMMA bf16) -> q.k^T (WMMA bf16) -> softmax ->
// per-score MLP (scalar->16->4, exact GELU) -> attention tensor + x_level.
// ---------------------------------------------------------------------------

typedef __attribute__((ext_vector_type(16))) __bf16 v16bf;
typedef __attribute__((ext_vector_type(8)))  float  v8f;

#define B_   4
#define TQ   1024
#define TK   1024
#define COV  64
#define TR   64
#define HID  16
#define OUTD 4

__device__ __forceinline__ float gelu_exact(float x) {
    // torch-default exact gelu: 0.5*x*(1+erf(x/sqrt(2)))
    return 0.5f * x * (1.0f + erff(x * 0.7071067811865475f));
}

// ---------------------------------------------------------------------------
// Kernel 1: fused LayerNorm + projection, one wave per 16-row tile.
// blockIdx.y: 0 -> (trend_query, ln_q_w, Wq) ; 1 -> (trend_key, ln_k_w, Wk)
// Output: bf16 rows (B*T, 64) into workspace.
// ---------------------------------------------------------------------------
__global__ __launch_bounds__(32) void ln_proj_kernel(
    const float* __restrict__ xq, const float* __restrict__ xk,
    const float* __restrict__ lnq, const float* __restrict__ lnk,
    const float* __restrict__ Wq,  const float* __restrict__ Wk,
    __bf16* __restrict__ qws, __bf16* __restrict__ kws)
{
    const int which = blockIdx.y;
    const float* x  = which ? xk  : xq;
    const float* lw = which ? lnk : lnq;
    const float* W  = which ? Wk  : Wq;
    __bf16* dst     = which ? kws : qws;

    const int g0   = blockIdx.x * 16;   // global row base (b*1024 + row)
    const int lane = threadIdx.x;       // 0..31
    const int m    = lane & 15;
    const int h    = lane >> 4;

    __shared__ __bf16 s_xn[16 * 64];
    __shared__ float  s_p[32], s_p2[32];
    __shared__ float  s_mu[16], s_rs[16];

    // ---- LayerNorm: 2 lanes per row, 32 elems each ----
    const float* row = x + (size_t)(g0 + m) * COV + h * 32;
    {
        float s = 0.f, s2 = 0.f;
        #pragma unroll
        for (int i = 0; i < 32; ++i) { float v = row[i]; s += v; s2 += v * v; }
        s_p[lane] = s; s_p2[lane] = s2;
        __syncthreads();
        if (h == 0) {
            float su = s_p[m] + s_p[m + 16];
            float sq = s_p2[m] + s_p2[m + 16];
            float mu = su * (1.0f / 64.0f);
            float var = sq * (1.0f / 64.0f) - mu * mu;   // population var (jnp.var)
            s_mu[m] = mu;
            s_rs[m] = rsqrtf(var + 1e-5f);
        }
        __syncthreads();
        float mu = s_mu[m], rs = s_rs[m];
        #pragma unroll
        for (int i = 0; i < 32; ++i) {
            int d = h * 32 + i;
            s_xn[m * 64 + d] = (__bf16)((row[i] - mu) * rs * lw[d]);
        }
        __syncthreads();
    }

    // ---- A fragments: 16x64 tile split into two 16x32 K-chunks ----
    // A layout (contiguous K, matching B): lane half h holds K=h*16..h*16+15
    // (elems 0..15) for row M=lane%16.
    v16bf a0, a1;
    #pragma unroll
    for (int i = 0; i < 8; ++i) {
        a0[i]     = s_xn[m * 64 +      h * 16 + i];
        a0[8 + i] = s_xn[m * 64 +  8 + h * 16 + i];
        a1[i]     = s_xn[m * 64 + 32 + h * 16 + i];
        a1[8 + i] = s_xn[m * 64 + 40 + h * 16 + i];
    }

    // ---- 4 N-tiles of the 64x64 weight, 2 WMMAs each (K=64) ----
    #pragma unroll
    for (int nt = 0; nt < 4; ++nt) {
        const int n = nt * 16 + m;
        v16bf wb0, wb1;   // B layout: lane holds column N=n, K split by lane half
        #pragma unroll
        for (int i = 0; i < 16; ++i) {
            wb0[i] = (__bf16)W[(h * 16 + i)      * TR + n];
            wb1[i] = (__bf16)W[(32 + h * 16 + i) * TR + n];
        }
        v8f c = {};
        c = __builtin_amdgcn_wmma_f32_16x16x32_bf16(false, a0, false, wb0, (short)0, c, false, false);
        c = __builtin_amdgcn_wmma_f32_16x16x32_bf16(false, a1, false, wb1, (short)0, c, false, false);
        // D layout: lane holds column N=n, rows M = 8*h + j
        #pragma unroll
        for (int j = 0; j < 8; ++j)
            dst[(size_t)(g0 + 8 * h + j) * TR + n] = (__bf16)c[j];
    }
}

// ---------------------------------------------------------------------------
// Kernel 2: fused scores (WMMA) + softmax + per-score MLP + outputs.
// One 256-thread workgroup per (batch, 16 q-rows). Scores live in LDS only.
// ---------------------------------------------------------------------------
__global__ __launch_bounds__(256) void attn_kernel(
    const __bf16* __restrict__ qws, const __bf16* __restrict__ kws,
    const float* __restrict__ tv,
    const float* __restrict__ w1, const float* __restrict__ b1,
    const float* __restrict__ w2, const float* __restrict__ b2,
    float* __restrict__ xout, float* __restrict__ attout)
{
    const int bid  = blockIdx.x;        // 0..255
    const int b    = bid >> 6;          // batch
    const int q0   = (bid & 63) * 16;   // q-row base within batch
    const int tid  = threadIdx.x;
    const int lane = tid & 31;
    const int wv   = tid >> 5;          // wave 0..7
    const int m    = lane & 15;
    const int h    = lane >> 4;

    __shared__ __align__(16) float s_sc[16 * 1024];   // 64 KB score tile
    __shared__ __bf16 s_q[16 * 64];
    __shared__ float  s_red[16 * 16];
    __shared__ float  s_max[16];
    __shared__ float  s_inv[16];
    __shared__ float  s_xl[16 * OUTD];

    // load q tile into LDS
    {
        const __bf16* src = qws + (size_t)(b * TQ + q0) * TR;
        for (int i = tid; i < 16 * 64; i += 256) s_q[i] = src[i];
    }
    if (tid < 16 * OUTD) s_xl[tid] = 0.f;
    __syncthreads();

    // A fragments (reused across all k tiles of this wave)
    v16bf a0, a1;
    #pragma unroll
    for (int i = 0; i < 8; ++i) {
        a0[i]     = s_q[m * 64 +      h * 16 + i];
        a0[8 + i] = s_q[m * 64 +  8 + h * 16 + i];
        a1[i]     = s_q[m * 64 + 32 + h * 16 + i];
        a1[8 + i] = s_q[m * 64 + 40 + h * 16 + i];
    }

    // scores: 64 k-tiles, wave wv owns tiles wv, wv+8, ...
    for (int kt = wv; kt < 64; kt += 8) {
        // B = k^T (64 x 16): column N = k-row (kt*16+m), contiguous bf16 row
        const __bf16* krow = kws + (size_t)(b * TK + kt * 16 + m) * TR;
        v16bf kb0, kb1;
        #pragma unroll
        for (int i = 0; i < 16; ++i) {
            kb0[i] = krow[h * 16 + i];
            kb1[i] = krow[32 + h * 16 + i];
        }
        v8f c = {};
        c = __builtin_amdgcn_wmma_f32_16x16x32_bf16(false, a0, false, kb0, (short)0, c, false, false);
        c = __builtin_amdgcn_wmma_f32_16x16x32_bf16(false, a1, false, kb1, (short)0, c, false, false);
        #pragma unroll
        for (int j = 0; j < 8; ++j)
            s_sc[(8 * h + j) * 1024 + kt * 16 + m] = c[j] * 0.125f;  // 1/sqrt(64)
    }
    __syncthreads();

    // ---- row softmax: 16 threads per row ----
    const int row = tid >> 4;
    const int sub = tid & 15;
    {
        float mx = -1e30f;
        for (int t = 0; t < 64; ++t)
            mx = fmaxf(mx, s_sc[row * 1024 + sub + 16 * t]);
        s_red[row * 16 + sub] = mx;
    }
    __syncthreads();
    if (sub == 0) {
        float mx = s_red[row * 16];
        #pragma unroll
        for (int i = 1; i < 16; ++i) mx = fmaxf(mx, s_red[row * 16 + i]);
        s_max[row] = mx;
    }
    __syncthreads();
    {
        const float mx = s_max[row];
        float sm = 0.f;
        for (int t = 0; t < 64; ++t) {
            int idx = row * 1024 + sub + 16 * t;
            float e = __expf(s_sc[idx] - mx);
            s_sc[idx] = e;
            sm += e;
        }
        s_red[row * 16 + sub] = sm;
    }
    __syncthreads();
    if (sub == 0) {
        float sm = 0.f;
        #pragma unroll
        for (int i = 0; i < 16; ++i) sm += s_red[row * 16 + i];
        s_inv[row] = 1.0f / sm;
    }
    __syncthreads();

    // ---- per-score MLP + outputs ----
    float rw1[HID], rb1[HID], rw2[OUTD][HID], rb2[OUTD];
    #pragma unroll
    for (int j = 0; j < HID; ++j) { rw1[j] = w1[j]; rb1[j] = b1[j]; }
    #pragma unroll
    for (int o = 0; o < OUTD; ++o) {
        rb2[o] = b2[o];
        #pragma unroll
        for (int j = 0; j < HID; ++j) rw2[o][j] = w2[o * HID + j];
    }
    float tvr[4];
    #pragma unroll
    for (int tt = 0; tt < 4; ++tt) tvr[tt] = tv[b * TK + tid + 256 * tt];

    // thread handles columns c = tid + 256*tt for every row r (coalesced writes)
    for (int r = 0; r < 16; ++r) {
        const float inv = s_inv[r];
        float xl[OUTD] = {0.f, 0.f, 0.f, 0.f};
        const size_t att_row_base =
            (size_t)b * OUTD * TQ * TK + (size_t)(q0 + r) * TK;
        #pragma unroll
        for (int tt = 0; tt < 4; ++tt) {
            const int c = tid + 256 * tt;
            const float att = s_sc[r * 1024 + c] * inv;
            float acc[OUTD] = {rb2[0], rb2[1], rb2[2], rb2[3]};
            #pragma unroll
            for (int j = 0; j < HID; ++j) {
                float hj = gelu_exact(att * rw1[j] + rb1[j]);
                acc[0] += rw2[0][j] * hj;
                acc[1] += rw2[1][j] * hj;
                acc[2] += rw2[2][j] * hj;
                acc[3] += rw2[3][j] * hj;
            }
            #pragma unroll
            for (int o = 0; o < OUTD; ++o) {
                attout[att_row_base + (size_t)o * TQ * TK + c] = acc[o];
                xl[o] += acc[o] * tvr[tt];
            }
        }
        #pragma unroll
        for (int o = 0; o < OUTD; ++o)
            atomicAdd(&s_xl[r * OUTD + o], xl[o]);   // ds_add_f32
    }
    __syncthreads();
    if (tid < 16 * OUTD) {
        int r = tid >> 2, o = tid & 3;
        xout[(size_t)(b * TQ + q0 + r) * OUTD + o] = s_xl[r * OUTD + o];
    }
}

// ---------------------------------------------------------------------------
extern "C" void kernel_launch(void* const* d_in, const int* in_sizes, int n_in,
                              void* d_out, int out_size, void* d_ws, size_t ws_size,
                              hipStream_t stream) {
    const float* tq  = (const float*)d_in[0];
    const float* tk  = (const float*)d_in[1];
    const float* tv  = (const float*)d_in[2];
    const float* lnq = (const float*)d_in[3];
    const float* lnk = (const float*)d_in[4];
    const float* Wq  = (const float*)d_in[5];
    const float* Wk  = (const float*)d_in[6];
    const float* w1  = (const float*)d_in[7];
    const float* b1  = (const float*)d_in[8];
    const float* w2  = (const float*)d_in[9];
    const float* b2  = (const float*)d_in[10];

    // workspace: projected q then k as bf16 rows (B*T, 64)
    __bf16* qws = (__bf16*)d_ws;
    __bf16* kws = (__bf16*)((char*)d_ws + (size_t)B_ * TQ * TR * sizeof(__bf16));

    float* xout   = (float*)d_out;                        // (B,Tq,OUT)
    float* attout = xout + (size_t)B_ * TQ * OUTD;        // (B,OUT,Tq,Tk)

    dim3 gA(B_ * TQ / 16, 2);
    ln_proj_kernel<<<gA, 32, 0, stream>>>(tq, tk, lnq, lnk, Wq, Wk, qws, kws);

    attn_kernel<<<B_ * (TQ / 16), 256, 0, stream>>>(
        qws, kws, tv, w1, b1, w2, b2, xout, attout);
}
